// MultiheadedAttention_32710470926954
// MI455X (gfx1250) — matrix operand order repaired and checked by reading, not compiled, not run on hardware
//
#include <hip/hip_runtime.h>
#include <hip/hip_bf16.h>
#include <math.h>

typedef __attribute__((ext_vector_type(16))) _Float16 v16h;
typedef __attribute__((ext_vector_type(8)))  float    v8f;
typedef __attribute__((ext_vector_type(4)))  int      v4i;

#define AS1 __attribute__((address_space(1)))
#define AS3 __attribute__((address_space(3)))

#define S_LEN 8192
#define DHEAD 64
#define WWID  512
#define NSPLIT 4
#define KSPAN (S_LEN / NSPLIT)   // 2048 keys per split

// ---------------------------------------------------------------------------
// gfx1250 async global->LDS copy (ASYNCcnt-tracked DMA), 16B per lane.
// ---------------------------------------------------------------------------
__device__ __forceinline__ void async_copy_b128(const void* gsrc, void* lds_dst) {
#if __has_builtin(__builtin_amdgcn_global_load_async_to_lds_b128)
    __builtin_amdgcn_global_load_async_to_lds_b128(
        (AS1 v4i*)(uintptr_t)gsrc,
        (AS3 v4i*)(uint32_t)(uintptr_t)lds_dst,
        0, 0);
#else
    uint32_t l = (uint32_t)(uintptr_t)lds_dst;
    asm volatile("global_load_async_to_lds_b128 %0, %1, off"
                 :: "v"(l), "v"(gsrc) : "memory");
#endif
}

__device__ __forceinline__ void wait_async0() {
#if __has_builtin(__builtin_amdgcn_s_wait_asynccnt)
    __builtin_amdgcn_s_wait_asynccnt(0);
#else
    asm volatile("s_wait_asynccnt 0" ::: "memory");
#endif
}

// Raw v_exp_f32: flushes exp2(x<-126) to 0, which is exactly what softmax wants.
__device__ __forceinline__ float fast_exp2(float x) {
#if __has_builtin(__builtin_amdgcn_exp2f)
    return __builtin_amdgcn_exp2f(x);
#else
    return exp2f(x);
#endif
}

// ---------------------------------------------------------------------------
// Prep 1: W_eff[n][d] = sum_h w_o[h*64+d][n], TRANSPOSED [512][64] f16.
// ---------------------------------------------------------------------------
__global__ __launch_bounds__(256) void reduce_wo_kernel(const float* __restrict__ w_o,
                                                        _Float16* __restrict__ Wt) {
    int t = blockIdx.x * 256 + threadIdx.x;
    int n = t >> 6;
    int d = t & 63;
    float s = 0.f;
#pragma unroll
    for (int h = 0; h < 8; ++h) s += w_o[(h * DHEAD + d) * WWID + n];
    Wt[n * DHEAD + d] = (_Float16)s;
}

// ---------------------------------------------------------------------------
// Prep 2: K -> f16 row-major; V -> f16 tile-transposed [S/32][64][32].
// ---------------------------------------------------------------------------
__global__ __launch_bounds__(256) void prep_kv_kernel(const float* __restrict__ K,
                                                      const float* __restrict__ V,
                                                      _Float16* __restrict__ Kh,
                                                      _Float16* __restrict__ Vt) {
    int i = blockIdx.x * 256 + threadIdx.x;       // 0 .. S*64-1
    int key = i >> 6;
    int dim = i & 63;
    Kh[i] = (_Float16)K[i];
    Vt[(key >> 5) * 2048 + dim * 32 + (key & 31)] = (_Float16)V[i];
}

// ---------------------------------------------------------------------------
// Pass 1: split-K flash attention partials. Grid (128, NSPLIT), 4 waves/block,
// 16 q-rows/wave, 32 keys/iter over a 2048-key span. Emits unnormalized O
// plus per-row (m, l). 2048 waves -> fills the machine.
// ---------------------------------------------------------------------------
__global__ __launch_bounds__(128) void attn_partial_kernel(const float* __restrict__ Q,
                                                           const _Float16* __restrict__ Kh,
                                                           const _Float16* __restrict__ Vt,
                                                           float* __restrict__ Opart,
                                                           float2* __restrict__ Ml) {
    __shared__ _Float16 Kbuf[2][2048];      // [key][dim] f16, 4 KB each
    __shared__ _Float16 Vbuf[2][2048];      // [dim][key] f16, 4 KB each

    const int tid  = threadIdx.x;
    const int lane = tid & 31;
    const int wave = tid >> 5;
    const int hh   = lane >> 4;
    const int l16  = lane & 15;
    const int qbase = blockIdx.x * 64 + wave * 16;
    const int split = blockIdx.y;
    const int kb0   = split * KSPAN;
    // 1/sqrt(64) * log2(e): scores in log2 domain -> exp2 softmax.
    const float scale = 0.125f * 1.44269504088896f;

    // Q^T as WMMA B-operand (2 chunks of headdim-32), loaded once.
    v16h qb[2];
#pragma unroll
    for (int c = 0; c < 2; ++c) {
        const float* qrow = Q + (qbase + l16) * DHEAD + 32 * c + 16 * hh;
#pragma unroll
        for (int j = 0; j < 8; ++j) {
            qb[c][2 * j]     = (_Float16)(qrow[2 * j]     * scale);
            qb[c][2 * j + 1] = (_Float16)(qrow[2 * j + 1] * scale);
        }
    }

    v8f acc[4] = {};
    float m_run = -INFINITY;
    float l_run = 0.f;

    // ---- prologue: async-fill buffer 0 with first tile ---------------------
#pragma unroll
    for (int r = 0; r < 2; ++r) {
        int off = tid * 16 + r * 2048;      // bytes; 128 lanes x 16B x 2 = 4 KB
        async_copy_b128((const char*)(Kh + kb0 * DHEAD) + off, (char*)Kbuf[0] + off);
        async_copy_b128((const char*)(Vt + (kb0 >> 5) * 2048) + off, (char*)Vbuf[0] + off);
    }
    wait_async0();
    __syncthreads();

#pragma unroll 1
    for (int kb = kb0; kb < kb0 + KSPAN; kb += 32) {
        const int cur = (kb >> 5) & 1;
        // ---- issue async DMA for the NEXT tile into the other buffer ------
        if (kb + 32 < kb0 + KSPAN) {
            const char* ksrc = (const char*)(Kh + (kb + 32) * DHEAD);
            const char* vsrc = (const char*)(Vt + ((kb + 32) >> 5) * 2048);
#pragma unroll
            for (int r = 0; r < 2; ++r) {
                int off = tid * 16 + r * 2048;
                async_copy_b128(ksrc + off, (char*)Kbuf[cur ^ 1] + off);
                async_copy_b128(vsrc + off, (char*)Vbuf[cur ^ 1] + off);
            }
        }

        // ---- S^T = K * Q^T ------------------------------------------------
        v16h ka[2][2];
#pragma unroll
        for (int t = 0; t < 2; ++t)
#pragma unroll
            for (int c = 0; c < 2; ++c) {
                const _Float16* krow = &Kbuf[cur][(16 * t + l16) * DHEAD + 32 * c + 16 * hh];
#pragma unroll
                for (int j = 0; j < 8; ++j) {
                    int kd = 2 * j;
                    ka[t][c][2 * j]     = krow[kd];
                    ka[t][c][2 * j + 1] = krow[kd + 1];
                }
            }

        v8f st[2];
#pragma unroll
        for (int t = 0; t < 2; ++t) {
            v8f s = {};
            s = __builtin_amdgcn_wmma_f32_16x16x32_f16(false, ka[t][0], false, qb[0],
                                                       (short)0, s, false, false);
            s = __builtin_amdgcn_wmma_f32_16x16x32_f16(false, ka[t][1], false, qb[1],
                                                       (short)0, s, false, false);
            st[t] = s;
        }

        // ---- prefetch V B-operands; latency hides under the softmax --------
        v16h vb[4];
#pragma unroll
        for (int dt = 0; dt < 4; ++dt) {
            const _Float16* vrow = &Vbuf[cur][(16 * dt + l16) * 32 + 8 * hh];
#pragma unroll
            for (int j = 0; j < 8; ++j) {
                vb[dt][2 * j]     = vrow[2 * j + 8 * (j >> 2)];
                vb[dt][2 * j + 1] = vrow[2 * j + 8 * (j >> 2) + 1];
            }
        }

        // ---- online softmax in exp2 domain ---------------------------------
        float mloc = st[0][0];
#pragma unroll
        for (int i = 0; i < 8; ++i) {
            mloc = fmaxf(mloc, st[0][i]);
            mloc = fmaxf(mloc, st[1][i]);
        }
        mloc = fmaxf(mloc, __shfl_xor(mloc, 16));
        float mnew   = fmaxf(m_run, mloc);
        float fscale = fast_exp2(m_run - mnew);
        float p0[8], p1[8];
        float lsum = 0.f;
#pragma unroll
        for (int i = 0; i < 8; ++i) {
            p0[i] = fast_exp2(st[0][i] - mnew);
            p1[i] = fast_exp2(st[1][i] - mnew);
            lsum += p0[i] + p1[i];
        }
        lsum += __shfl_xor(lsum, 16);
        l_run = l_run * fscale + lsum;
        m_run = mnew;

        // ---- rescale accumulator only when the running max moved -----------
        if (__any(fscale < 1.0f)) {
#pragma unroll
            for (int i = 0; i < 8; ++i) {
                float fi = __shfl(fscale, i + 8 * hh);
#pragma unroll
                for (int t = 0; t < 4; ++t) acc[t][i] *= fi;
            }
        }

        // ---- pack P (transposed scores == A-layout) -------------------------
        v16h pa;
#pragma unroll
        for (int j = 0; j < 4; ++j) {
            pa[2 * j]         = (_Float16)p0[2 * j];
            pa[2 * j + 1]     = (_Float16)p0[2 * j + 1];
            pa[8 + 2 * j]     = (_Float16)p1[2 * j];
            pa[8 + 2 * j + 1] = (_Float16)p1[2 * j + 1];
        }

        // ---- O += P * V -----------------------------------------------------
#pragma unroll
        for (int dt = 0; dt < 4; ++dt)
            acc[dt] = __builtin_amdgcn_wmma_f32_16x16x32_f16(false, pa, false, vb[dt],
                                                             (short)0, acc[dt], false, false);

        wait_async0();
        __syncthreads();
    }

    // ---- emit unnormalized partial O + per-row (m, l) ----------------------
    float* op = Opart + ((size_t)split * S_LEN + qbase) * DHEAD;
#pragma unroll
    for (int t = 0; t < 4; ++t)
#pragma unroll
        for (int i = 0; i < 8; ++i)
            op[(i + 8 * hh) * DHEAD + 16 * t + l16] = acc[t][i];
    if (lane < 16) {
        float2 ml; ml.x = m_run; ml.y = l_run;
        Ml[split * S_LEN + qbase + lane] = ml;
    }
}

// ---------------------------------------------------------------------------
// Pass 2: softmax-combine the NSPLIT partials + output projection.
// A-layout row index == lane index, so the combine is fully lane-local.
// ---------------------------------------------------------------------------
__global__ __launch_bounds__(128) void proj_combine_kernel(const float* __restrict__ Opart,
                                                           const float2* __restrict__ Ml,
                                                           const _Float16* __restrict__ Wt,
                                                           float* __restrict__ out) {
    const int lane = threadIdx.x & 31;
    const int wave = threadIdx.x >> 5;
    const int hh   = lane >> 4;
    const int l16  = lane & 15;
    const int qbase = blockIdx.x * 64 + wave * 16;
    const int row   = qbase + l16;

    // Per-row combine weights: w_s = exp2(m_s - m*) / sum_s exp2(m_s - m*) l_s
    float e[NSPLIT], l_s[NSPLIT];
    float mstar = -INFINITY;
#pragma unroll
    for (int s = 0; s < NSPLIT; ++s) {
        float2 ml = Ml[s * S_LEN + row];
        e[s]   = ml.x;
        l_s[s] = ml.y;
        mstar  = fmaxf(mstar, ml.x);
    }
    float L = 0.f;
#pragma unroll
    for (int s = 0; s < NSPLIT; ++s) {
        e[s] = fast_exp2(e[s] - mstar);
        L += e[s] * l_s[s];
    }
    float rL = 1.0f / L;
#pragma unroll
    for (int s = 0; s < NSPLIT; ++s) e[s] *= rL;

    // Build O as WMMA A-operand directly from global partials (lane-local).
    v16h oa[2];
#pragma unroll
    for (int c = 0; c < 2; ++c)
#pragma unroll
        for (int j = 0; j < 8; ++j) {
            int dd = 32 * c + 2 * j + 16 * hh;
            float a0 = 0.f, a1 = 0.f;
#pragma unroll
            for (int s = 0; s < NSPLIT; ++s) {
                const float* op = Opart + ((size_t)s * S_LEN + row) * DHEAD;
                a0 += e[s] * op[dd];
                a1 += e[s] * op[dd + 1];
            }
            oa[c][2 * j]     = (_Float16)a0;
            oa[c][2 * j + 1] = (_Float16)a1;
        }

    // ---- projection: out[16 x 512] = O(16x64) @ W_eff(64x512) --------------
#pragma unroll 4
    for (int nt = 0; nt < 32; ++nt) {
        v8f o = {};
#pragma unroll
        for (int c = 0; c < 2; ++c) {
            v16h wb;
            const _Float16* wrow = Wt + (nt * 16 + l16) * DHEAD + 32 * c + 16 * hh;
#pragma unroll
            for (int j = 0; j < 8; ++j) {
                wb[2 * j]     = wrow[2 * j];
                wb[2 * j + 1] = wrow[2 * j + 1];
            }
            o = __builtin_amdgcn_wmma_f32_16x16x32_f16(false, oa[c], false, wb,
                                                       (short)0, o, false, false);
        }
#pragma unroll
        for (int i = 0; i < 8; ++i)
            out[(qbase + i + 8 * hh) * WWID + nt * 16 + l16] = o[i];
    }
}

extern "C" void kernel_launch(void* const* d_in, const int* in_sizes, int n_in,
                              void* d_out, int out_size, void* d_ws, size_t ws_size,
                              hipStream_t stream) {
    const float* q   = (const float*)d_in[0];
    const float* k   = (const float*)d_in[1];
    const float* v   = (const float*)d_in[2];
    const float* w_o = (const float*)d_in[3];
    float* out = (float*)d_out;

    // Workspace: Wt 64KB | Kh 1MB | Vt 1MB | Opart 8MB | Ml 256KB  (~10.4 MB)
    char* ws = (char*)d_ws;
    _Float16* Wt    = (_Float16*)ws;
    _Float16* Kh    = (_Float16*)(ws + (64 << 10));
    _Float16* Vt    = (_Float16*)(ws + (64 << 10) + S_LEN * DHEAD * 2);
    float*    Opart = (float*)   (ws + (64 << 10) + 2 * S_LEN * DHEAD * 2);
    float2*   Ml    = (float2*)  (ws + (64 << 10) + 2 * S_LEN * DHEAD * 2
                                     + (size_t)NSPLIT * S_LEN * DHEAD * 4);

    reduce_wo_kernel<<<(WWID * DHEAD) / 256, 256, 0, stream>>>(w_o, Wt);
    prep_kv_kernel<<<(S_LEN * DHEAD) / 256, 256, 0, stream>>>(k, v, Kh, Vt);
    attn_partial_kernel<<<dim3(S_LEN / 64, NSPLIT), 128, 0, stream>>>(q, Kh, Vt, Opart, Ml);
    proj_combine_kernel<<<S_LEN / 64, 128, 0, stream>>>(Opart, Ml, Wt, out);
}
